// NysAttention_52029233823829
// MI455X (gfx1250) — compile-verified, matrix operand order checked by reading, not run on hardware
//
#include <hip/hip_runtime.h>
#include <hip/hip_bf16.h>
#include <stdint.h>

#define NHEADS 16
#define NLAND  64
#define DIMC   1024
#define HD     64
#define NB     2
#define NN     8192
#define BHC    (NB*NHEADS)   // 32
#define MROWS  (NB*NN)       // 16384

typedef __attribute__((ext_vector_type(16))) __bf16 v16bf;
typedef __attribute__((ext_vector_type(8)))  float  v8f;
typedef __attribute__((ext_vector_type(4)))  int    v4i;

#if defined(__has_builtin)
#  if __has_builtin(__builtin_amdgcn_global_load_async_to_lds_b128) && \
      __has_builtin(__builtin_amdgcn_s_wait_asynccnt)
#    define ASYNC_LDS 1
#  endif
#endif
#ifndef ASYNC_LDS
#  define ASYNC_LDS 0
#endif

__device__ __forceinline__ unsigned short f2bf(float f){
  uint32_t u = __float_as_uint(f);
  u += 0x7FFFu + ((u >> 16) & 1u);          // round-to-nearest-even
  return (unsigned short)(u >> 16);
}
__device__ __forceinline__ float bf2f(unsigned short h){
  return __uint_as_float(((uint32_t)h) << 16);
}

union Frag16 { unsigned short u[16]; v16bf v; };

// 16x32 bf16 WMMA fragment load. A and B share the lane map when the source is
// stored [outer][k] row-major (outer = M-row for A, N-col for B).
// Lane map (ISA 7.12.2): lane%16 = outer, lane/16 picks the K-halves;
// VGPR r<4: K = 8*khi + 2r + h ; r>=4: K = 16 + 8*khi + 2(r-4) + h.
__device__ __forceinline__ v16bf load_frag16(const unsigned short* __restrict__ base,
                                             int stride, int outerBase, int kBase, int lane){
  Frag16 f;
  const unsigned short* p = base + (size_t)(outerBase + (lane & 15)) * stride + kBase;
  int khi = (lane >> 4) << 3;   // 0 or 8
  #pragma unroll
  for (int r = 0; r < 4; ++r){
    f.u[2*r]       = p[khi + 2*r];
    f.u[2*r + 1]   = p[khi + 2*r + 1];
    f.u[8 + 2*r]   = p[16 + khi + 2*r];
    f.u[8 + 2*r+1] = p[16 + khi + 2*r + 1];
  }
  return f.v;
}

__device__ __forceinline__ v8f wmma_bf16(v16bf a, v16bf b, v8f c){
  return __builtin_amdgcn_wmma_f32_16x16x32_bf16(false, a, false, b, (short)0, c, false, false);
}

// 16B/lane global -> LDS copy: async path (ASYNCcnt) when available,
// else regs round-trip.
__device__ __forceinline__ void copy16B_to_lds(unsigned short* lds_dst,
                                               const unsigned short* gsrc){
#if ASYNC_LDS
  __builtin_amdgcn_global_load_async_to_lds_b128(
      (__attribute__((address_space(1))) v4i*)gsrc,
      (__attribute__((address_space(3))) v4i*)lds_dst,
      0, 0);
#else
  *(uint4*)lds_dst = *(const uint4*)gsrc;
#endif
}
__device__ __forceinline__ void async_copies_wait(){
#if ASYNC_LDS
  __builtin_amdgcn_s_wait_asynccnt(0);
#endif
}

// ---------------------------------------------------------------- converts
__global__ void f32_to_bf16_kernel(const float* __restrict__ src,
                                   unsigned short* __restrict__ dst, int n){
  int i = blockIdx.x * blockDim.x + threadIdx.x;
  int stride = gridDim.x * blockDim.x;
  for (; i < n; i += stride) dst[i] = f2bf(src[i]);
}

// ---------------------------------------------------------------- big GEMM
// C[M,Nc] = A[M,K]bf16 @ B[K,Nc]bf16.
// Block tile 128x128, wave tile 32x64, K step 32, double-buffered LDS,
// one barrier per K step, async A-tile fill overlapped with WMMA.
// mode 0: split qkv epilogue (scale q by 1/8, scatter to [B,H,N,64] bf16)
// mode 1: plain fp32 store
#define TM 128
#define TN 128
#define TK 32
__global__ __launch_bounds__(256) void gemm_bf16_kernel(
    const unsigned short* __restrict__ A, const unsigned short* __restrict__ Bm,
    int M, int K, int Ncols, int mode,
    float* __restrict__ Cout,
    unsigned short* __restrict__ qb, unsigned short* __restrict__ kb,
    unsigned short* __restrict__ vb)
{
  __shared__ __align__(16) unsigned short sA[2][TM][TK];   // 2 x 8 KB
  __shared__ __align__(16) unsigned short sB[2][TN][TK];   // 2 x 8 KB, [col][k]
  int tid = threadIdx.x, lane = tid & 31, wave = tid >> 5;
  int nBase = blockIdx.x * TN;
  int mBase = blockIdx.y * TM;
  int wm = (wave & 3) * 32;     // wave row offset within tile
  int wn = (wave >> 2) * 64;    // wave col offset within tile

  v8f acc[2][4];
  #pragma unroll
  for (int mi = 0; mi < 2; ++mi)
    #pragma unroll
    for (int ni = 0; ni < 4; ++ni)
      #pragma unroll
      for (int j = 0; j < 8; ++j) acc[mi][ni][j] = 0.0f;

  // ---- stage K-chunk 0 into buffer 0
  #pragma unroll
  for (int c = 0; c < 2; ++c){                 // A: 512 x 16B chunks
    int ch = tid + c*256;
    int row = ch >> 2, off = ch & 3;
    copy16B_to_lds(&sA[0][row][off*8], A + (size_t)(mBase+row)*K + off*8);
  }
  #pragma unroll
  for (int i = 0; i < 8; ++i){                 // B: transpose into [col][k]
    int d = tid + i*256;
    int kk = d >> 6, cu = d & 63;
    uint32_t w = *(const uint32_t*)(Bm + (size_t)kk*Ncols + nBase + cu*2);
    sB[0][cu*2][kk]   = (unsigned short)(w & 0xFFFFu);
    sB[0][cu*2+1][kk] = (unsigned short)(w >> 16);
  }
  async_copies_wait();
  __syncthreads();

  int cur = 0;
  for (int k0 = 0; k0 < K; k0 += TK){
    int nxt = cur ^ 1;
    bool has_next = (k0 + TK < K);
    uint32_t bpre[8];
    if (has_next){
      // issue next A-tile async copies + next B-tile global loads first,
      // so they overlap the WMMA work below
      #pragma unroll
      for (int c = 0; c < 2; ++c){
        int ch = tid + c*256;
        int row = ch >> 2, off = ch & 3;
        copy16B_to_lds(&sA[nxt][row][off*8],
                       A + (size_t)(mBase+row)*K + (k0 + TK) + off*8);
      }
      #pragma unroll
      for (int i = 0; i < 8; ++i){
        int d = tid + i*256;
        int kk = d >> 6, cu = d & 63;
        bpre[i] = *(const uint32_t*)(Bm + (size_t)(k0 + TK + kk)*Ncols + nBase + cu*2);
      }
    }

    // ---- compute current chunk: 8 WMMA per wave
    v16bf af0 = load_frag16(&sA[cur][0][0], TK, wm,      0, lane);
    v16bf af1 = load_frag16(&sA[cur][0][0], TK, wm + 16, 0, lane);
    #pragma unroll
    for (int ni = 0; ni < 4; ++ni){
      v16bf bf = load_frag16(&sB[cur][0][0], TK, wn + ni*16, 0, lane);
      acc[0][ni] = wmma_bf16(af0, bf, acc[0][ni]);
      acc[1][ni] = wmma_bf16(af1, bf, acc[1][ni]);
    }

    if (has_next){
      #pragma unroll
      for (int i = 0; i < 8; ++i){
        int d = tid + i*256;
        int kk = d >> 6, cu = d & 63;
        sB[nxt][cu*2][kk]   = (unsigned short)(bpre[i] & 0xFFFFu);
        sB[nxt][cu*2+1][kk] = (unsigned short)(bpre[i] >> 16);
      }
      async_copies_wait();
    }
    __syncthreads();
    cur = nxt;
  }

  int rOff = (lane >> 4) ? 8 : 0;
  #pragma unroll
  for (int mi = 0; mi < 2; ++mi){
    #pragma unroll
    for (int t = 0; t < 4; ++t){
      int col = nBase + wn + t*16 + (lane & 15);
      #pragma unroll
      for (int r = 0; r < 8; ++r){
        int row = mBase + wm + mi*16 + rOff + r;
        float val = acc[mi][t][r];
        if (mode == 1){
          Cout[(size_t)row * Ncols + col] = val;
        } else {
          int which = col >> 10;        // 0=q 1=k 2=v
          int hc = col & 1023;
          int head = hc >> 6, d = hc & 63;
          int b = row >> 13, n = row & 8191;
          size_t off = (((size_t)(b*NHEADS + head))*NN + n)*HD + d;
          unsigned short* dst = (which == 0) ? qb : (which == 1) ? kb : vb;
          dst[off] = f2bf(which == 0 ? val * 0.125f : val);   // q /= sqrt(64)
        }
      }
    }
  }
}

// ------------------------------------------------------- landmark pooling
__global__ void pool_kernel(const unsigned short* __restrict__ qb,
                            const unsigned short* __restrict__ kb,
                            float* __restrict__ qlm, float* __restrict__ klm,
                            unsigned short* __restrict__ klmb)
{
  int bh = blockIdx.y, l = blockIdx.x, d = threadIdx.x;    // 64 threads
  const unsigned short* qp = qb + (((size_t)bh*NN) + l*128)*HD + d;
  const unsigned short* kp = kb + (((size_t)bh*NN) + l*128)*HD + d;
  float sq = 0.f, sk = 0.f;
  for (int t = 0; t < 128; ++t){ sq += bf2f(qp[(size_t)t*HD]); sk += bf2f(kp[(size_t)t*HD]); }
  float mq = sq * (1.0f/128.0f), mk = sk * (1.0f/128.0f);
  size_t o = ((size_t)bh*NLAND + l)*HD + d;
  qlm[o] = mq; klm[o] = mk; klmb[o] = f2bf(mk);
}

// ---------------------------------------- kernel3 softmax fused with @ v
__global__ __launch_bounds__(256) void k3v_fused_kernel(
    const float* __restrict__ qlm, const unsigned short* __restrict__ kb,
    const unsigned short* __restrict__ vb, float* __restrict__ k3v)
{
  __shared__ float sc[NN];
  __shared__ float red[256];
  __shared__ float sq[HD];
  int bh = blockIdx.y, l = blockIdx.x, tid = threadIdx.x;
  if (tid < HD) sq[tid] = qlm[((size_t)bh*NLAND + l)*HD + tid];
  __syncthreads();

  const unsigned short* kball = kb + (size_t)bh*NN*HD;
  for (int i = 0; i < 32; ++i){
    int n = tid + i*256;
    const unsigned short* kr = kball + (size_t)n*HD;
    float s = 0.f;
    #pragma unroll 8
    for (int d = 0; d < HD; d += 2){
      uint32_t two = *(const uint32_t*)(kr + d);
      s += sq[d]   * bf2f((unsigned short)(two & 0xFFFFu));
      s += sq[d+1] * bf2f((unsigned short)(two >> 16));
    }
    sc[n] = s;
  }
  __syncthreads();

  float mx = -3.0e38f;
  for (int i = 0; i < 32; ++i) mx = fmaxf(mx, sc[tid + i*256]);
  red[tid] = mx; __syncthreads();
  for (int s = 128; s > 0; s >>= 1){
    if (tid < s) red[tid] = fmaxf(red[tid], red[tid+s]);
    __syncthreads();
  }
  float rowmax = red[0]; __syncthreads();

  float sum = 0.f;
  for (int i = 0; i < 32; ++i){
    int n = tid + i*256;
    float e = __expf(sc[n] - rowmax);
    sc[n] = e; sum += e;
  }
  red[tid] = sum; __syncthreads();
  for (int s = 128; s > 0; s >>= 1){
    if (tid < s) red[tid] += red[tid+s];
    __syncthreads();
  }
  float tot = red[0]; __syncthreads();

  int d = tid & 63, g = tid >> 6;                 // 4 n-groups x 64 dims
  const unsigned short* vball = vb + (size_t)bh*NN*HD;
  float accv = 0.f;
  for (int n = g; n < NN; n += 4) accv += sc[n] * bf2f(vball[(size_t)n*HD + d]);
  red[tid] = accv; __syncthreads();
  if (tid < HD){
    float r = red[tid] + red[tid+64] + red[tid+128] + red[tid+192];
    k3v[((size_t)bh*NLAND + l)*HD + tid] = r / tot;
  }
}

// ---------------- kernel2 softmax + Newton-Schulz inverse + W2 = inv @ k3v
__device__ __forceinline__ void mm64_regs(const float* A, const float* B,
                                          float* r16, int tid){
  #pragma unroll
  for (int e0 = 0; e0 < 16; ++e0){
    int e = tid*16 + e0; int i = e >> 6, j = e & 63;
    float s = 0.f;
    #pragma unroll 8
    for (int k = 0; k < 64; ++k) s += A[i*64+k] * B[k*64+j];
    r16[e0] = s;
  }
}

__global__ __launch_bounds__(256) void nys_inv_kernel(
    const float* __restrict__ qlm, const float* __restrict__ klm,
    const float* __restrict__ k3v, unsigned short* __restrict__ w2t)
{
  __shared__ float Ksm[4096], Vsm[4096], Csm[4096], Dsm[4096];  // 64 KB
  int bh = blockIdx.x, tid = threadIdx.x;
  const float* ql = qlm + (size_t)bh*4096;
  const float* kl = klm + (size_t)bh*4096;
  float r16[16];

  #pragma unroll
  for (int e0 = 0; e0 < 16; ++e0){             // S = q_lm @ k_lm^T
    int e = tid*16 + e0; int i = e >> 6, j = e & 63;
    float s = 0.f;
    for (int d = 0; d < 64; ++d) s += ql[i*64+d] * kl[j*64+d];
    Ksm[e] = s;
  }
  __syncthreads();
  if (tid < 64){                               // fp32 row softmax
    float mx = -3.0e38f;
    for (int j = 0; j < 64; ++j) mx = fmaxf(mx, Ksm[tid*64+j]);
    float sum = 0.f;
    for (int j = 0; j < 64; ++j){ float e = __expf(Ksm[tid*64+j]-mx); Ksm[tid*64+j]=e; sum+=e; }
    float inv = 1.0f/sum;
    for (int j = 0; j < 64; ++j) Ksm[tid*64+j] *= inv;
  }
  __syncthreads();
  if (tid < 64){ float cs = 0.f; for (int i = 0; i < 64; ++i) cs += Ksm[i*64+tid]; Dsm[tid] = cs; }
  __syncthreads();
  if (tid == 0){ float m = Dsm[0]; for (int j = 1; j < 64; ++j) m = fmaxf(m, Dsm[j]); Dsm[64] = m; }
  __syncthreads();
  float denom = Dsm[64];
  __syncthreads();
  #pragma unroll
  for (int e0 = 0; e0 < 16; ++e0){             // V = K^T / denom
    int e = tid*16 + e0; int i = e >> 6, j = e & 63;
    Vsm[e] = Ksm[j*64+i] / denom;
  }
  __syncthreads();

  for (int it = 0; it < 6; ++it){
    mm64_regs(Ksm, Vsm, r16, tid);  __syncthreads();        // C = K@V
    #pragma unroll
    for (int e0 = 0; e0 < 16; ++e0) Csm[tid*16+e0] = r16[e0];
    __syncthreads();
    #pragma unroll
    for (int e0 = 0; e0 < 16; ++e0){                        // D = 7I - C
      int e = tid*16 + e0; int i = e >> 6, j = e & 63;
      Dsm[e] = ((i == j) ? 7.0f : 0.0f) - Csm[e];
    }
    __syncthreads();
    mm64_regs(Csm, Dsm, r16, tid);  __syncthreads();        // D = 15I - C@D
    #pragma unroll
    for (int e0 = 0; e0 < 16; ++e0){
      int e = tid*16 + e0; int i = e >> 6, j = e & 63;
      Dsm[e] = ((i == j) ? 15.0f : 0.0f) - r16[e0];
    }
    __syncthreads();
    mm64_regs(Csm, Dsm, r16, tid);  __syncthreads();        // D = 13I - C@D
    #pragma unroll
    for (int e0 = 0; e0 < 16; ++e0){
      int e = tid*16 + e0; int i = e >> 6, j = e & 63;
      Dsm[e] = ((i == j) ? 13.0f : 0.0f) - r16[e0];
    }
    __syncthreads();
    mm64_regs(Vsm, Dsm, r16, tid);  __syncthreads();        // V = 0.25 V@D
    #pragma unroll
    for (int e0 = 0; e0 < 16; ++e0) Vsm[tid*16+e0] = 0.25f * r16[e0];
    __syncthreads();
  }

  const float* z = k3v + (size_t)bh*4096;       // W2^T bf16 [d][j]
  #pragma unroll
  for (int e0 = 0; e0 < 16; ++e0){
    int e = tid*16 + e0; int j = e >> 6, d = e & 63;
    float s = 0.f;
    for (int l = 0; l < 64; ++l) s += Vsm[j*64+l] * z[l*64+d];
    w2t[(size_t)bh*4096 + d*64 + j] = f2bf(s);
  }
}

// ------------------- kernel1 = softmax(q @ k_lm^T) fused, WMMA + shfl softmax
__global__ __launch_bounds__(256) void kernel1_softmax_kernel(
    const unsigned short* __restrict__ qb, const unsigned short* __restrict__ klmb,
    unsigned short* __restrict__ k1b)
{
  __shared__ unsigned short sKlm[64*64];    // [mcol][d], 8 KB
  int bh = blockIdx.y;
  int rowBase = blockIdx.x * 128;
  int tid = threadIdx.x, lane = tid & 31, wave = tid >> 5;

  const uint32_t* src = (const uint32_t*)(klmb + (size_t)bh*4096);
  #pragma unroll
  for (int i = 0; i < 8; ++i) ((uint32_t*)sKlm)[tid + i*256] = src[tid + i*256];
  __syncthreads();

  const unsigned short* arow = qb + ((size_t)bh*NN + rowBase + wave*16)*HD;
  v8f acc[4];
  #pragma unroll
  for (int t = 0; t < 4; ++t)
    #pragma unroll
    for (int j = 0; j < 8; ++j) acc[t][j] = 0.0f;

  #pragma unroll
  for (int kc = 0; kc < 2; ++kc){
    v16bf af = load_frag16(arow, HD, 0, kc*32, lane);
    #pragma unroll
    for (int t = 0; t < 4; ++t){
      v16bf bf = load_frag16(sKlm, HD, t*16, kc*32, lane);
      acc[t] = wmma_bf16(af, bf, acc[t]);
    }
  }

  #pragma unroll
  for (int r = 0; r < 8; ++r){     // row softmax over 64 cols (16-lane halves)
    float mx = -3.0e38f;
    #pragma unroll
    for (int t = 0; t < 4; ++t) mx = fmaxf(mx, acc[t][r]);
    #pragma unroll
    for (int s = 8; s >= 1; s >>= 1) mx = fmaxf(mx, __shfl_xor(mx, s, 32));
    float sum = 0.f;
    #pragma unroll
    for (int t = 0; t < 4; ++t){ float e = __expf(acc[t][r] - mx); acc[t][r] = e; sum += e; }
    #pragma unroll
    for (int s = 8; s >= 1; s >>= 1) sum += __shfl_xor(sum, s, 32);
    float inv = 1.0f / sum;
    #pragma unroll
    for (int t = 0; t < 4; ++t) acc[t][r] *= inv;
  }

  unsigned short* dst = k1b + ((size_t)bh*NN + rowBase + wave*16)*HD;
  int colL = lane & 15, rOff = (lane >> 4) ? 8 : 0;
  #pragma unroll
  for (int t = 0; t < 4; ++t)
    #pragma unroll
    for (int r = 0; r < 8; ++r)
      dst[(size_t)(rOff + r)*HD + t*16 + colL] = f2bf(acc[t][r]);
}

// ---------------------------- SV = kernel1 @ W2 ; dual fp32/bf16 store
__global__ __launch_bounds__(256) void sv_gemm_kernel(
    const unsigned short* __restrict__ k1b, const unsigned short* __restrict__ w2t,
    float* __restrict__ svout, unsigned short* __restrict__ svb)
{
  __shared__ unsigned short sW[64*64];      // [d][j] transposed W2, 8 KB
  int bh = blockIdx.y;
  int rowBase = blockIdx.x * 128;
  int b = bh >> 4, h = bh & 15;
  int tid = threadIdx.x, lane = tid & 31, wave = tid >> 5;

  const uint32_t* src = (const uint32_t*)(w2t + (size_t)bh*4096);
  #pragma unroll
  for (int i = 0; i < 8; ++i) ((uint32_t*)sW)[tid + i*256] = src[tid + i*256];
  __syncthreads();

  const unsigned short* arow = k1b + ((size_t)bh*NN + rowBase + wave*16)*HD;
  v8f acc[4];
  #pragma unroll
  for (int t = 0; t < 4; ++t)
    #pragma unroll
    for (int j = 0; j < 8; ++j) acc[t][j] = 0.0f;

  #pragma unroll
  for (int kc = 0; kc < 2; ++kc){
    v16bf af = load_frag16(arow, HD, 0, kc*32, lane);
    #pragma unroll
    for (int t = 0; t < 4; ++t){
      v16bf bf = load_frag16(sW, HD, t*16, kc*32, lane);
      acc[t] = wmma_bf16(af, bf, acc[t]);
    }
  }

  int rOff = (lane >> 4) ? 8 : 0;
  #pragma unroll
  for (int t = 0; t < 4; ++t){
    int col = t*16 + (lane & 15);
    #pragma unroll
    for (int r = 0; r < 8; ++r){
      int n = rowBase + wave*16 + rOff + r;
      size_t o = ((size_t)(b*NN + n))*DIMC + h*HD + col;   // [B,N,C] layout
      float v = acc[t][r];
      svout[o] = v;
      svb[o]  = f2bf(v);
    }
  }
}

// ---------------------------------------------------------------- launch
extern "C" void kernel_launch(void* const* d_in, const int* in_sizes, int n_in,
                              void* d_out, int out_size, void* d_ws, size_t ws_size,
                              hipStream_t stream) {
  (void)in_sizes; (void)n_in; (void)out_size; (void)ws_size;
  const float* x      = (const float*)d_in[0];   // [B,N,C]
  const float* w_qkv  = (const float*)d_in[1];   // [C,3C]
  const float* w_proj = (const float*)d_in[2];   // [C,C]

  float* out   = (float*)d_out;                          // [B,N,C]
  float* svout = out + (size_t)MROWS * DIMC;             // [B,N,C]

  size_t off = 0;
  auto alloc = [&](size_t bytes) -> char* {
    char* p = (char*)d_ws + off;
    off += (bytes + 255) & ~(size_t)255;
    return p;
  };
  unsigned short* xb     = (unsigned short*)alloc((size_t)MROWS*DIMC*2);     // reused as k1b
  unsigned short* wqkvb  = (unsigned short*)alloc((size_t)DIMC*3*DIMC*2);
  unsigned short* wprojb = (unsigned short*)alloc((size_t)DIMC*DIMC*2);
  unsigned short* qb     = (unsigned short*)alloc((size_t)BHC*NN*HD*2);      // reused as svb
  unsigned short* kb     = (unsigned short*)alloc((size_t)BHC*NN*HD*2);
  unsigned short* vb     = (unsigned short*)alloc((size_t)BHC*NN*HD*2);
  float*          qlm    = (float*)alloc((size_t)BHC*NLAND*HD*4);
  float*          klm    = (float*)alloc((size_t)BHC*NLAND*HD*4);
  unsigned short* klmb   = (unsigned short*)alloc((size_t)BHC*NLAND*HD*2);
  float*          k3v    = (float*)alloc((size_t)BHC*NLAND*HD*4);
  unsigned short* w2tb   = (unsigned short*)alloc((size_t)BHC*NLAND*HD*2);
  unsigned short* k1b = xb;        // x dead after qkv GEMM
  unsigned short* svb = qb;        // q dead after pooling + kernel1

  // 1. fp32 -> bf16 operand conversion
  f32_to_bf16_kernel<<<4096, 256, 0, stream>>>(x,      xb,     MROWS*DIMC);
  f32_to_bf16_kernel<<<4096, 256, 0, stream>>>(w_qkv,  wqkvb,  DIMC*3*DIMC);
  f32_to_bf16_kernel<<<2048, 256, 0, stream>>>(w_proj, wprojb, DIMC*DIMC);

  // 2. qkv = x @ w_qkv  (103 GF, WMMA, double-buffered + async LDS fill)
  gemm_bf16_kernel<<<dim3(3*DIMC/TN, MROWS/TM), 256, 0, stream>>>(
      xb, wqkvb, MROWS, DIMC, 3*DIMC, 0, nullptr, qb, kb, vb);

  // 3. landmark means
  pool_kernel<<<dim3(NLAND, BHC), 64, 0, stream>>>(qb, kb, qlm, klm, klmb);

  // 4. k3v = softmax(q_lm @ k^T) @ v   (fused, scores row in LDS)
  k3v_fused_kernel<<<dim3(NLAND, BHC), 256, 0, stream>>>(qlm, kb, vb, k3v);

  // 5. W2 = NewtonSchulz(softmax(q_lm @ k_lm^T))^-1 @ k3v   (fp32 LDS)
  nys_inv_kernel<<<BHC, 256, 0, stream>>>(qlm, klm, k3v, w2tb);

  // 6. kernel1 = softmax(q @ k_lm^T)   (WMMA + fused row softmax)
  kernel1_softmax_kernel<<<dim3(NN/128, BHC), 256, 0, stream>>>(qb, klmb, k1b);

  // 7. SV = kernel1 @ W2   (WMMA; stores fp32 SV output + bf16 copy)
  sv_gemm_kernel<<<dim3(NN/128, BHC), 256, 0, stream>>>(k1b, w2tb, svout, svb);

  // 8. out = SV @ w_proj   (34 GF, WMMA)
  gemm_bf16_kernel<<<dim3(DIMC/TN, MROWS/TM), 256, 0, stream>>>(
      svb, wprojb, MROWS, DIMC, DIMC, 1, out, nullptr, nullptr, nullptr);
}
